// LinearAttention_8022998909473
// MI455X (gfx1250) — compile-verified, matrix operand order checked by reading, not run on hardware
//
#include <hip/hip_runtime.h>

// Problem constants (B=4, N=4096, DIM=512, HEADS=8, DIM_HEAD=64)
#define BB      4
#define NN      4096
#define DIMK    512
#define HEADS   8
#define DH      64
#define EOUT    1536              // 3 * HEADS * DH
#define MROWS   (BB * NN)         // 16384
#define PANEL   (BB * HEADS * NN * DH)   // 8388608 floats per q/k/v panel

typedef float  v2f  __attribute__((ext_vector_type(2)));
typedef float  v8f  __attribute__((ext_vector_type(8)));
typedef __bf16 v16bf __attribute__((ext_vector_type(16)));
typedef __bf16 v8bf  __attribute__((ext_vector_type(8)));

// ---------------------------------------------------------------------------
// Kernel 1: qkv = x @ W + b via bf16x3 split-precision WMMA (K=32/instr),
// fp32 accumulate. Block tile 128(M) x 64(N), BK = 64. 8 waves, each 32x32.
// Scatter into q/k/v panels laid out (B,H,N,64).
// ---------------------------------------------------------------------------
#define BM   128
#define BN   64
#define BK   64
#define LDAB 72   // bf16 stride: 144 B rows -> 16B-aligned frag chunks

__global__ __launch_bounds__(256) void qkv_gemm_kernel(
    const float* __restrict__ x, const float* __restrict__ W,
    const float* __restrict__ bias,
    float* __restrict__ q, float* __restrict__ k, float* __restrict__ v)
{
    __shared__ __align__(16) __bf16 Ah[BM * LDAB];   // 18432 B
    __shared__ __align__(16) __bf16 Al[BM * LDAB];   // 18432 B
    __shared__ __align__(16) __bf16 BTh[BN * LDAB];  //  9216 B (transposed: [e][k])
    __shared__ __align__(16) __bf16 BTl[BN * LDAB];  //  9216 B

    const int tid   = threadIdx.x;
    const int lane  = tid & 31;
    const int w     = tid >> 5;
    const int wm    = w & 3;          // M offset wm*32
    const int wn    = w >> 2;         // N offset wn*32
    const int l16   = lane & 15;
    const int lhalf = lane >> 4;

    const int rowBase = blockIdx.x * BM;   // gridDim.x = 128
    const int colBase = blockIdx.y * BN;   // gridDim.y = 24

    v8f acc[2][2] = {};

    for (int kb = 0; kb < DIMK; kb += BK) {
        // ---- stage A: 128 x 64 fp32 -> bf16 hi/lo ----
        {
            const int c4 = (tid & 15) * 4;    // 16 float4 per row of 64
            const int r0 = tid >> 4;          // 16 rows per pass
            #pragma unroll
            for (int p = 0; p < 8; ++p) {
                const int r = r0 + 16 * p;
                const float4 f = *(const float4*)(x + (size_t)(rowBase + r) * DIMK + kb + c4);
                __bf16* dh = &Ah[r * LDAB + c4];
                __bf16* dl = &Al[r * LDAB + c4];
                const float fv[4] = {f.x, f.y, f.z, f.w};
                #pragma unroll
                for (int t = 0; t < 4; ++t) {
                    const __bf16 hi = (__bf16)fv[t];
                    dh[t] = hi;
                    dl[t] = (__bf16)(fv[t] - (float)hi);
                }
            }
        }
        // ---- stage B: 64(K) x 64(E) fp32 -> bf16 hi/lo, transposed [e][k] ----
        {
            const int e    = tid & 63;
            const int rgrp = tid >> 6;        // 0..3
            #pragma unroll
            for (int p = 0; p < 16; ++p) {
                const int r = rgrp + 4 * p;   // K row 0..63
                const float f  = W[(size_t)(kb + r) * EOUT + colBase + e];
                const __bf16 hi = (__bf16)f;
                BTh[e * LDAB + r] = hi;
                BTl[e * LDAB + r] = (__bf16)(f - (float)hi);
            }
        }
        // prefetch next A tile (gfx1250 global_prefetch_b8 path)
        if (kb + BK < DIMK) {
            __builtin_prefetch(x + (size_t)(rowBase + (tid >> 4)) * DIMK + kb + BK, 0, 1);
        }
        __syncthreads();

        #pragma unroll
        for (int ks = 0; ks < 2; ++ks) {       // two K=32 steps per stage
            const int base = ks * 32 + 8 * lhalf;
            v16bf ah[2], al[2], bh[2], bl[2];
            #pragma unroll
            for (int i = 0; i < 2; ++i) {
                const int row = wm * 32 + i * 16 + l16;
                const v8bf h0 = *(const v8bf*)&Ah[row * LDAB + base];
                const v8bf h1 = *(const v8bf*)&Ah[row * LDAB + base + 16];
                const v8bf l0 = *(const v8bf*)&Al[row * LDAB + base];
                const v8bf l1 = *(const v8bf*)&Al[row * LDAB + base + 16];
                #pragma unroll
                for (int t = 0; t < 8; ++t) {
                    ah[i][t] = h0[t]; ah[i][t + 8] = h1[t];
                    al[i][t] = l0[t]; al[i][t + 8] = l1[t];
                }
            }
            #pragma unroll
            for (int j = 0; j < 2; ++j) {
                const int col = wn * 32 + j * 16 + l16;
                const v8bf h0 = *(const v8bf*)&BTh[col * LDAB + base];
                const v8bf h1 = *(const v8bf*)&BTh[col * LDAB + base + 16];
                const v8bf l0 = *(const v8bf*)&BTl[col * LDAB + base];
                const v8bf l1 = *(const v8bf*)&BTl[col * LDAB + base + 16];
                #pragma unroll
                for (int t = 0; t < 8; ++t) {
                    bh[j][t] = h0[t]; bh[j][t + 8] = h1[t];
                    bl[j][t] = l0[t]; bl[j][t + 8] = l1[t];
                }
            }
            #pragma unroll
            for (int i = 0; i < 2; ++i) {
                #pragma unroll
                for (int j = 0; j < 2; ++j) {
                    acc[i][j] = __builtin_amdgcn_wmma_f32_16x16x32_bf16(
                        false, ah[i], false, bh[j], (short)0, acc[i][j], false, false);
                    acc[i][j] = __builtin_amdgcn_wmma_f32_16x16x32_bf16(
                        false, ah[i], false, bl[j], (short)0, acc[i][j], false, false);
                    acc[i][j] = __builtin_amdgcn_wmma_f32_16x16x32_bf16(
                        false, al[i], false, bh[j], (short)0, acc[i][j], false, false);
                }
            }
        }
        __syncthreads();
    }

    // ---- epilogue: add bias, scatter to (B,H,N,64) panels ----
    #pragma unroll
    for (int i = 0; i < 2; ++i) {
        #pragma unroll
        for (int j = 0; j < 2; ++j) {
            const int ecol = colBase + wn * 32 + j * 16 + l16;   // 0..1535
            const float bb = bias[ecol];
            const int slot = ecol >> 6;        // 0..23
            const int part = slot >> 3;        // 0=q 1=k 2=v
            const int head = slot & 7;
            const int d    = ecol & 63;
            float* dst = (part == 0) ? q : (part == 1) ? k : v;
            #pragma unroll
            for (int r = 0; r < 8; ++r) {
                const int m  = rowBase + wm * 32 + i * 16 + r + 8 * lhalf;
                const int bi = m >> 12;          // m / 4096
                const int n  = m & (NN - 1);
                dst[(((size_t)(bi * HEADS + head) * NN + n) << 6) + d] = acc[i][j][r] + bb;
            }
        }
    }
}

// ---------------------------------------------------------------------------
// Kernel 2: per-(b,h,d) masked online max/sum over n for k-softmax (axis=-2).
// ---------------------------------------------------------------------------
__global__ __launch_bounds__(256) void kcol_stats_kernel(
    const float* __restrict__ k, const unsigned char* __restrict__ mask,
    float* __restrict__ Mb, float* __restrict__ Zi)
{
    __shared__ float ms[256];
    __shared__ float ss[256];

    const int bh  = blockIdx.x;          // 0..31
    const int b   = bh >> 3;
    const int tid = threadIdx.x;
    const int d   = tid & 63;
    const int g   = tid >> 6;            // 0..3

    const float* kp = k + (size_t)bh * NN * DH;
    const unsigned char* mp = mask + (size_t)b * NN;

    float m = -3.402823466e38f;
    float s = 0.f;
    for (int n = g; n < NN; n += 4) {
        if (mp[n]) {
            const float xv = kp[(size_t)n * DH + d];
            if (xv > m) { s = s * __expf(m - xv) + 1.f; m = xv; }
            else        { s += __expf(xv - m); }
        }
    }
    ms[tid] = m;
    ss[tid] = s;
    __syncthreads();

    if (g == 0) {
        float M = ms[d], S = ss[d];
        #pragma unroll
        for (int jj = 1; jj < 4; ++jj) {
            const float m2 = ms[d + 64 * jj];
            const float s2 = ss[d + 64 * jj];
            if (m2 > M) { S = S * __expf(M - m2) + s2; M = m2; }
            else        { S += s2 * __expf(m2 - M); }
        }
        Mb[bh * DH + d] = M;
        Zi[bh * DH + d] = (S > 0.f) ? (1.f / S) : 0.f;
    }
}

// ---------------------------------------------------------------------------
// Kernel 2b: zero the kv accumulator region (needed every launch).
// ---------------------------------------------------------------------------
__global__ __launch_bounds__(256) void kv_zero_kernel(float* __restrict__ kv)
{
    const int i = (blockIdx.x * 256 + threadIdx.x) * 4;   // 128 blocks -> 131072
    kv[i + 0] = 0.f; kv[i + 1] = 0.f; kv[i + 2] = 0.f; kv[i + 3] = 0.f;
}

// ---------------------------------------------------------------------------
// Kernel 3: kv[bh][d][e] = sum_n exp(k[n,d]-M_d)*mask[n] * v[n,e]
// fp32 WMMA (precision-critical exp products). Partials via atomic_add_f32.
// ---------------------------------------------------------------------------
#define LDT 68   // kT stride (float2 frag loads conflict-free)
#define LDV 72   // v stride  (half-wave rows 16 banks apart)

__global__ __launch_bounds__(256) void kv_gemm_kernel(
    const float* __restrict__ k, const float* __restrict__ v,
    const unsigned char* __restrict__ mask, const float* __restrict__ Mb,
    float* __restrict__ kv)
{
    __shared__ float kTs[DH * LDT];   // 17408 B
    __shared__ float vs[DH * LDV];    // 18432 B

    const int bh    = blockIdx.x;            // 0..31
    const int b     = bh >> 3;
    const int chunk = blockIdx.y;            // 0..7 (512 rows each)
    const int tid   = threadIdx.x;
    const int lane  = tid & 31;
    const int w     = tid >> 5;
    const int l16   = lane & 15;
    const int lhalf = lane >> 4;
    const int dt    = w >> 1;                // d-tile 0..3
    const int e0    = (w & 1) * 2;           // e-tiles e0, e0+1

    const int   dcol = tid & 63;
    const int   rgrp = tid >> 6;             // 0..3
    const float Mloc = Mb[bh * DH + dcol];

    const float* kp = k + (size_t)bh * NN * DH;
    const float* vp = v + (size_t)bh * NN * DH;
    const unsigned char* mp = mask + (size_t)b * NN;

    v8f acc[2] = {};

    for (int n0 = chunk * 512; n0 < chunk * 512 + 512; n0 += 64) {
        #pragma unroll
        for (int p = 0; p < 16; ++p) {
            const int r  = rgrp + 4 * p;
            const int n  = n0 + r;
            const float mk = mp[n] ? 1.f : 0.f;
            const float kvv = kp[(size_t)n * DH + dcol];
            kTs[dcol * LDT + r] = mk * __expf(kvv - Mloc);
            vs[r * LDV + dcol]  = vp[(size_t)n * DH + dcol];
        }
        __syncthreads();

        #pragma unroll
        for (int kk = 0; kk < 64; kk += 4) {
            v2f a;
            {
                const float2 ap = *(const float2*)&kTs[(dt * 16 + l16) * LDT + kk + 2 * lhalf];
                a.x = ap.x; a.y = ap.y;
            }
            #pragma unroll
            for (int j = 0; j < 2; ++j) {
                v2f bfr;
                const int col = (e0 + j) * 16 + l16;
                const int rk  = kk + 2 * lhalf;
                bfr.x = vs[rk * LDV + col];
                bfr.y = vs[(rk + 1) * LDV + col];
                acc[j] = __builtin_amdgcn_wmma_f32_16x16x4_f32(
                    false, a, false, bfr, (short)0, acc[j], false, false);
            }
        }
        __syncthreads();
    }

    float* kvb = kv + (size_t)bh * DH * DH;
    #pragma unroll
    for (int j = 0; j < 2; ++j) {
        const int e = (e0 + j) * 16 + l16;
        #pragma unroll
        for (int r = 0; r < 8; ++r) {
            const int d = dt * 16 + r + 8 * lhalf;
            atomicAdd(&kvb[d * DH + e], acc[j][r]);
        }
    }
}

// ---------------------------------------------------------------------------
// Kernel 4: out[b][n][h*64+e] = sum_d softmax_row(q)[n,d] * (kv[d,e]/Z_d)
// ---------------------------------------------------------------------------
#define LDQ 68   // q stride
#define LDKV 72  // kv stride

__global__ __launch_bounds__(256) void out_gemm_kernel(
    const float* __restrict__ q, const float* __restrict__ kv,
    const float* __restrict__ Zi, float* __restrict__ out)
{
    __shared__ float kvs[DH * LDKV];    // 18432 B
    __shared__ float qs[128 * LDQ];     // 34816 B

    const int bh    = blockIdx.x;        // 0..31
    const int b     = bh >> 3;
    const int h     = bh & 7;
    const int n0    = blockIdx.y * 128;  // 0..4095 step 128
    const int tid   = threadIdx.x;
    const int lane  = tid & 31;
    const int w     = tid >> 5;          // wave owns rows 16w..16w+15
    const int l16   = lane & 15;
    const int lhalf = lane >> 4;

    const float* qp  = q  + (size_t)bh * NN * DH;
    const float* kvp = kv + (size_t)bh * DH * DH;

    #pragma unroll
    for (int p = 0; p < 16; ++p) {
        const int idx = tid + 256 * p;   // 0..4095
        const int d = idx >> 6, e = idx & 63;
        kvs[d * LDKV + e] = kvp[idx] * Zi[bh * DH + d];
    }
    {
        const int dcol = tid & 63;
        const int rgrp = tid >> 6;
        #pragma unroll
        for (int p = 0; p < 32; ++p) {
            const int r = rgrp + 4 * p;
            qs[r * LDQ + dcol] = qp[(size_t)(n0 + r) * DH + dcol];
        }
    }
    __syncthreads();

    if (tid < 128) {
        float mx = -3.402823466e38f;
        #pragma unroll 8
        for (int d = 0; d < DH; ++d) mx = fmaxf(mx, qs[tid * LDQ + d]);
        float s = 0.f;
        #pragma unroll 8
        for (int d = 0; d < DH; ++d) {
            const float e = __expf(qs[tid * LDQ + d] - mx);
            qs[tid * LDQ + d] = e;
            s += e;
        }
        const float inv = 1.f / s;
        #pragma unroll 8
        for (int d = 0; d < DH; ++d) qs[tid * LDQ + d] *= inv;
    }
    __syncthreads();

    v8f acc[4] = {};
    #pragma unroll
    for (int kk = 0; kk < 64; kk += 4) {
        v2f a;
        {
            const float2 ap = *(const float2*)&qs[(w * 16 + l16) * LDQ + kk + 2 * lhalf];
            a.x = ap.x; a.y = ap.y;
        }
        #pragma unroll
        for (int j = 0; j < 4; ++j) {
            v2f bfr;
            const int col = j * 16 + l16;
            const int rk  = kk + 2 * lhalf;
            bfr.x = kvs[rk * LDKV + col];
            bfr.y = kvs[(rk + 1) * LDKV + col];
            acc[j] = __builtin_amdgcn_wmma_f32_16x16x4_f32(
                false, a, false, bfr, (short)0, acc[j], false, false);
        }
    }

    #pragma unroll
    for (int j = 0; j < 4; ++j) {
        const int e = j * 16 + l16;
        #pragma unroll
        for (int r = 0; r < 8; ++r) {
            const int n = n0 + w * 16 + r + 8 * lhalf;
            out[((size_t)(b * NN + n)) * (HEADS * DH) + h * DH + e] = acc[j][r];
        }
    }
}

// ---------------------------------------------------------------------------
// Launch
// ---------------------------------------------------------------------------
extern "C" void kernel_launch(void* const* d_in, const int* in_sizes, int n_in,
                              void* d_out, int out_size, void* d_ws, size_t ws_size,
                              hipStream_t stream)
{
    const float*         x    = (const float*)d_in[0];
    const unsigned char* mask = (const unsigned char*)d_in[1];  // jnp bool -> 1 byte
    const float*         W    = (const float*)d_in[2];
    const float*         bias = (const float*)d_in[3];
    float*               out  = (float*)d_out;

    float* ws = (float*)d_ws;
    float* q  = ws;                 // PANEL floats
    float* k  = q + PANEL;
    float* v  = k + PANEL;
    float* kv = v + PANEL;          // 32*64*64 = 131072 floats
    float* Mb = kv + 32 * DH * DH;  // 2048
    float* Zi = Mb + 32 * DH;       // 2048

    qkv_gemm_kernel <<<dim3(MROWS / BM, EOUT / BN), 256, 0, stream>>>(x, W, bias, q, k, v);
    kcol_stats_kernel<<<32, 256, 0, stream>>>(k, mask, Mb, Zi);
    kv_zero_kernel  <<<128, 256, 0, stream>>>(kv);
    kv_gemm_kernel  <<<dim3(32, 8), 256, 0, stream>>>(k, v, mask, Mb, kv);
    out_gemm_kernel <<<dim3(32, 32), 256, 0, stream>>>(q, kv, Zi, out);
}
